// CrossModalFuzzyAttention_10814727651431
// MI455X (gfx1250) — matrix operand order repaired and checked by reading, not compiled, not run on hardware
//
#include <hip/hip_runtime.h>
#include <hip/hip_bf16.h>

typedef __attribute__((ext_vector_type(16))) __bf16 v16bf;
typedef __attribute__((ext_vector_type(8)))  float  v8f;
typedef unsigned short ushort_t;

static __device__ __forceinline__ ushort_t f2bf(float f) {
    unsigned int u = __float_as_uint(f);
    u += 0x7FFFu + ((u >> 16) & 1u);            // round-to-nearest-even
    return (ushort_t)(u >> 16);
}

// packed f32x2 -> bf16x2 (one V_CVT_PK_BF16_F32 when available)
#if __has_builtin(__builtin_amdgcn_cvt_pk_bf16_f32)
typedef __attribute__((ext_vector_type(2))) __bf16 v2bf;
static __device__ __forceinline__ unsigned int f2bf_pk(float lo, float hi) {
    union { v2bf v; unsigned int u; } r;
    r.v = __builtin_amdgcn_cvt_pk_bf16_f32(lo, hi);
    return r.u;
}
#else
static __device__ __forceinline__ unsigned int f2bf_pk(float lo, float hi) {
    return (unsigned int)f2bf(lo) | ((unsigned int)f2bf(hi) << 16);
}
#endif

// convert a pair with one packed cvt, then split into two b16 stores
static __device__ __forceinline__ void store_bf16_pair(ushort_t* p0, ushort_t* p1,
                                                       float a, float b) {
    unsigned int pk = f2bf_pk(a, b);
    *p0 = (ushort_t)pk;
    *p1 = (ushort_t)(pk >> 16);
}

union AFrag {
    v16bf v;
    ushort_t u[16];
    uint4 q[2];
};

union U8x16 {
    uint4 q;
    ushort_t u[8];
};

static __device__ __forceinline__ v8f wmma_bf16(const AFrag& a, const AFrag& b, v8f c) {
    return __builtin_amdgcn_wmma_f32_16x16x32_bf16(
        /*neg_a=*/false, a.v, /*neg_b=*/false, b.v,
        /*c_mod=*/(short)0, c, /*reuse_a=*/false, /*reuse_b=*/false);
}

// LDS byte address for async-to-LDS: generic addr low 32 bits == LDS offset (ISA §10.2)
static __device__ __forceinline__ unsigned int lds_addr32(const void* p) {
    return (unsigned int)(uintptr_t)p;
}

// CDNA5 async DMA: LDS[vdst] = MEM[vaddr], 16B per lane, tracked by ASYNCcnt
static __device__ __forceinline__ void async_load_b128(unsigned int lds, const void* gaddr) {
    asm volatile("global_load_async_to_lds_b128 %0, %1, off"
                 :: "v"(lds), "v"(gaddr) : "memory");
}
static __device__ __forceinline__ void wait_async0() {
    asm volatile("s_wait_asynccnt 0x0" ::: "memory");
}

// ---------------------------------------------------------------- convert ----
__global__ void conv_f32_bf16(const float* __restrict__ in, ushort_t* __restrict__ out, int n2) {
    int i = blockIdx.x * blockDim.x + threadIdx.x;
    int stride = gridDim.x * blockDim.x;
    for (; i < n2; i += stride) {                       // n2 = n/2, all sizes even
        float2 v = *(const float2*)(in + 2 * (size_t)i);
        *(unsigned int*)(out + 2 * (size_t)i) = f2bf_pk(v.x, v.y);
    }
}

// ------------------------------------------------------------------- GEMM ----
// C[M,N] = epilogue(A_bf16[M,K] @ W_f32[K,N] + bias)
// Workgroup: 256 threads = 8 waves; tile 64(M) x 128(N); K chunks of 32.
enum { MODE_NONE = 0, MODE_SIGMOID = 1, MODE_RESID = 2 };

template <int MODE>
__global__ __launch_bounds__(256)
void gemm_bf16(const ushort_t* __restrict__ A, const float* __restrict__ W,
               const float* __restrict__ bias, ushort_t* __restrict__ Cb,
               float* __restrict__ Cf, const float* __restrict__ resid,
               const float* __restrict__ fwp, int M, int N, int K) {
    __shared__ __align__(16) ushort_t Alds[64][40];   // [m][k], pad 8 (row stride 80B)
    __shared__ __align__(16) ushort_t Wt[128][40];    // [n][k] transposed, pad 8

    const int t    = threadIdx.x;
    const int lane = t & 31, wave = t >> 5;
    const int hi   = lane >> 4, ln = lane & 15;
    const int mt   = wave & 3, nt = wave >> 2;        // wave tile: 16 rows x 64 cols
    const int m0   = blockIdx.x * 64;
    const int n0   = blockIdx.y * 128;

    // per-thread staging coordinates (loop-invariant)
    const int arow = t >> 2, aseg = t & 3;            // A: 64 rows x 4 x 8-elem segs
    const int wkp = t >> 4, wcs = t & 15;             // W: 16 k-pairs x 16 col-segs(8)
    const unsigned int a_lds = lds_addr32(&Alds[arow][8 * aseg]);

    v8f acc[4] = {};

    for (int kc = 0; kc < K; kc += 32) {
        // A chunk [64 x 32] bf16: async DMA straight into LDS (16B per lane)
        async_load_b128(a_lds, A + (size_t)(m0 + arow) * K + kc + 8 * aseg);

        // W chunk [32 x 128] f32 -> bf16, transposed, packed pairs along k
        {
            const float* w0 = W + (size_t)(kc + 2 * wkp) * N + n0 + 8 * wcs;
            const float* w1 = w0 + N;
#pragma unroll
            for (int i = 0; i < 8; ++i)
                *(unsigned int*)(&Wt[8 * wcs + i][2 * wkp]) = f2bf_pk(w0[i], w1[i]);
        }
        wait_async0();
        __syncthreads();

        // A operand (ISA layout: lane<16 -> K 0-7,16-23; lane>=16 -> K 8-15,24-31)
        AFrag a;
        a.q[0] = *(const uint4*)(&Alds[16 * mt + ln][hi * 8]);
        a.q[1] = *(const uint4*)(&Alds[16 * mt + ln][16 + hi * 8]);
#pragma unroll
        for (int j = 0; j < 4; ++j) {
            AFrag b;                                   // lane = column, K order matches A (0-7,16-23 / 8-15,24-31)
            int col = 64 * nt + 16 * j + ln;
            b.q[0] = *(const uint4*)(&Wt[col][hi * 8]);
            b.q[1] = *(const uint4*)(&Wt[col][16 + hi * 8]);
            acc[j] = wmma_bf16(a, b, acc[j]);
        }
        __syncthreads();
    }

    float fwv = 0.0f;
    if (MODE == MODE_RESID) fwv = 1.0f / (1.0f + __expf(-fwp[0]));

#pragma unroll
    for (int j = 0; j < 4; ++j) {
        int col = n0 + 64 * nt + 16 * j + ln;
        float bv = bias[col];
        float v[8];
#pragma unroll
        for (int r = 0; r < 8; ++r) {                  // C layout: M = r + 8*hi, N = ln
            v[r] = acc[j][r] + bv;
            if (MODE == MODE_SIGMOID) v[r] = 1.0f / (1.0f + __expf(-v[r]));
        }
        const int row0 = m0 + 16 * mt + 8 * hi;
        if (MODE == MODE_RESID) {
#pragma unroll
            for (int r = 0; r < 8; ++r) {
                size_t idx = (size_t)(row0 + r) * N + col;
                Cf[idx] = (1.0f - fwv) * resid[idx] + fwv * v[r];
            }
        } else {
#pragma unroll
            for (int r = 0; r < 8; r += 2)             // packed cvt, split b16 stores
                store_bf16_pair(Cb + (size_t)(row0 + r) * N + col,
                                Cb + (size_t)(row0 + r + 1) * N + col,
                                v[r], v[r + 1]);
        }
    }
}

// -------------------------------------------------------- fused attention ----
// Q,K in (0,1) (sigmoid memberships), scale 1/8 => scores in [0,8]:
// softmax needs no running max. Row sums via extra WMMA against all-ones B.
// grid = (B*H, S/64); block = 128 (4 waves, 16 q-rows each).
__global__ __launch_bounds__(128)
void fuzzy_attn(const ushort_t* __restrict__ Qg, const ushort_t* __restrict__ Kg,
                const ushort_t* __restrict__ Vg, ushort_t* __restrict__ Og) {
    __shared__ __align__(16) ushort_t Kch[32][72];     // [key][feat], pad 8 (row 144B)
    __shared__ __align__(16) ushort_t Vt[64][40];      // [feat][key] transposed, pad 8
    __shared__ __align__(16) ushort_t Pl[4][16][32];   // per-wave P bounce (C->A relayout)

    const int t     = threadIdx.x;
    const int lane  = t & 31, wave = t >> 5;
    const int hi    = lane >> 4, ln = lane & 15;
    const int batch = blockIdx.x >> 2, head = blockIdx.x & 3;
    const int rb    = batch * 1024;
    const int q0    = rb + blockIdx.y * 64 + wave * 16;
    const int c0    = head * 64;

    // per-thread staging coordinates
    const int kkey = t >> 2, kseg = t & 3;             // K: 32 keys x 2 x b128
    const int vkp  = t & 15, vfs = t >> 4;             // V: 16 key-pairs x 8 feat-segs(8)
    const unsigned int k_lds0 = lds_addr32(&Kch[kkey][16 * kseg]);
    const unsigned int k_lds1 = lds_addr32(&Kch[kkey][16 * kseg + 8]);

    // Q tile (16 x 64) as two A operands (feats 0-31, 32-63)
    AFrag qa0, qa1;
    {
        const ushort_t* qp = Qg + (size_t)(q0 + ln) * 256 + c0;
        qa0.q[0] = *(const uint4*)(qp + hi * 8);
        qa0.q[1] = *(const uint4*)(qp + 16 + hi * 8);
        qa1.q[0] = *(const uint4*)(qp + 32 + hi * 8);
        qa1.q[1] = *(const uint4*)(qp + 48 + hi * 8);
    }
    AFrag ones;
#pragma unroll
    for (int i = 0; i < 16; ++i) ones.u[i] = 0x3F80;   // bf16 1.0

    v8f Oa[4] = {};
    v8f L = {};

    for (int kc = 0; kc < 1024; kc += 32) {
        // K chunk [32 x 64]: async DMA into LDS (two b128 per lane)
        {
            const ushort_t* kp = Kg + (size_t)(rb + kc + kkey) * 256 + c0 + 16 * kseg;
            async_load_b128(k_lds0, kp);
            async_load_b128(k_lds1, kp + 8);
        }
        // V chunk transposed: key-pairs packed into b32 LDS stores
        {
            const ushort_t* v0 = Vg + (size_t)(rb + kc + 2 * vkp) * 256 + c0 + 8 * vfs;
            U8x16 r0, r1;
            r0.q = *(const uint4*)(v0);
            r1.q = *(const uint4*)(v0 + 256);
#pragma unroll
            for (int i = 0; i < 8; ++i)
                *(unsigned int*)(&Vt[8 * vfs + i][2 * vkp]) =
                    (unsigned int)r0.u[i] | ((unsigned int)r1.u[i] << 16);
        }
        wait_async0();
        __syncthreads();

        // scores (two 16x16 key tiles, contraction over dh=64) -> exp -> Pl
#pragma unroll
        for (int kt = 0; kt < 2; ++kt) {
            AFrag b0, b1;
            int key = 16 * kt + ln;
            b0.q[0] = *(const uint4*)(&Kch[key][hi * 8]);
            b0.q[1] = *(const uint4*)(&Kch[key][16 + hi * 8]);
            b1.q[0] = *(const uint4*)(&Kch[key][32 + hi * 8]);
            b1.q[1] = *(const uint4*)(&Kch[key][48 + hi * 8]);
            v8f s = {};
            s = wmma_bf16(qa0, b0, s);
            s = wmma_bf16(qa1, b1, s);
            float p[8];
#pragma unroll
            for (int r = 0; r < 8; ++r)
                p[r] = __expf(s[r] * 0.125f);          // bounded: s in [0,8]
#pragma unroll
            for (int r = 0; r < 8; r += 2)             // packed cvt, split b16 stores
                store_bf16_pair(&Pl[wave][r + 8 * hi][key],
                                &Pl[wave][r + 1 + 8 * hi][key],
                                p[r], p[r + 1]);
        }
        // per-wave LDS RAW: stores above -> loads below
        asm volatile("s_wait_dscnt 0x0" ::: "memory");

        AFrag pa;                                      // P as A operand (16 x 32 keys)
        pa.q[0] = *(const uint4*)(&Pl[wave][ln][hi * 8]);
        pa.q[1] = *(const uint4*)(&Pl[wave][ln][16 + hi * 8]);
#pragma unroll
        for (int j = 0; j < 4; ++j) {                  // O += P @ V  (4 feat tiles)
            AFrag vb;
            int col = 16 * j + ln;
            vb.q[0] = *(const uint4*)(&Vt[col][hi * 8]);
            vb.q[1] = *(const uint4*)(&Vt[col][16 + hi * 8]);
            Oa[j] = wmma_bf16(pa, vb, Oa[j]);
        }
        L = wmma_bf16(pa, ones, L);                    // row sums, lane-local with Oa
        __syncthreads();
    }

#pragma unroll
    for (int j = 0; j < 4; ++j) {
        const int col  = c0 + 16 * j + ln;
        const int row0 = q0 + 8 * hi;
        float o[8];
#pragma unroll
        for (int r = 0; r < 8; ++r) o[r] = Oa[j][r] / L[r];
#pragma unroll
        for (int r = 0; r < 8; r += 2)
            store_bf16_pair(Og + (size_t)(row0 + r) * 256 + col,
                            Og + (size_t)(row0 + r + 1) * 256 + col,
                            o[r], o[r + 1]);
    }
}

// ----------------------------------------------------------------- launch ----
extern "C" void kernel_launch(void* const* d_in, const int* in_sizes, int n_in,
                              void* d_out, int out_size, void* d_ws, size_t ws_size,
                              hipStream_t stream) {
    (void)in_sizes; (void)n_in; (void)out_size; (void)ws_size;

    const float* tf  = (const float*)d_in[0];   // [8,1024,512]
    const float* imf = (const float*)d_in[1];   // [8,1024,768]
    const float* Wtp = (const float*)d_in[2];  const float* btp = (const float*)d_in[3];
    const float* Wip = (const float*)d_in[4];  const float* bip = (const float*)d_in[5];
    const float* Wq  = (const float*)d_in[6];  const float* bq  = (const float*)d_in[7];
    const float* Wk  = (const float*)d_in[8];  const float* bk  = (const float*)d_in[9];
    const float* Wv  = (const float*)d_in[10]; const float* bv  = (const float*)d_in[11];
    const float* Wo  = (const float*)d_in[12]; const float* bo  = (const float*)d_in[13];
    const float* Wto = (const float*)d_in[14]; const float* bto = (const float*)d_in[15];
    const float* Wio = (const float*)d_in[16]; const float* bio = (const float*)d_in[17];
    const float* fw  = (const float*)d_in[18];

    const int M = 8192;                          // B*S flattened
    char* ws = (char*)d_ws;
    size_t off = 0;
    ushort_t* tf16  = (ushort_t*)(ws + off); off += (size_t)M * 512 * 2;   //  8 MB
    ushort_t* if16  = (ushort_t*)(ws + off); off += (size_t)M * 768 * 2;   // 12 MB
    ushort_t* tproj = (ushort_t*)(ws + off); off += (size_t)M * 256 * 2;   //  4 MB (reused as E2)
    ushort_t* iproj = (ushort_t*)(ws + off); off += (size_t)M * 256 * 2;
    ushort_t* Qb    = (ushort_t*)(ws + off); off += (size_t)M * 256 * 2;
    ushort_t* Kb    = (ushort_t*)(ws + off); off += (size_t)M * 256 * 2;
    ushort_t* Vb    = (ushort_t*)(ws + off); off += (size_t)M * 256 * 2;
    ushort_t* Ob    = (ushort_t*)(ws + off); off += (size_t)M * 256 * 2;   // O1 then O2
    ushort_t* E1    = (ushort_t*)(ws + off); off += (size_t)M * 256 * 2;
    ushort_t* E2    = tproj;                    // tproj dead once Q2/K2/V2 are built

    float* outT = (float*)d_out;
    float* outI = (float*)d_out + (size_t)M * 512;

    dim3 blk(256);

    conv_f32_bf16<<<2048, 256, 0, stream>>>(tf,  tf16, M * 512 / 2);
    conv_f32_bf16<<<2048, 256, 0, stream>>>(imf, if16, M * 768 / 2);

    // projections into common hidden space
    gemm_bf16<MODE_NONE><<<dim3(M / 64, 2), blk, 0, stream>>>(tf16, Wtp, btp, tproj, nullptr, nullptr, nullptr, M, 256, 512);
    gemm_bf16<MODE_NONE><<<dim3(M / 64, 2), blk, 0, stream>>>(if16, Wip, bip, iproj, nullptr, nullptr, nullptr, M, 256, 768);

    // attention 1: text queries, image keys/values
    gemm_bf16<MODE_SIGMOID><<<dim3(M / 64, 2), blk, 0, stream>>>(tproj, Wq, bq, Qb, nullptr, nullptr, nullptr, M, 256, 256);
    gemm_bf16<MODE_SIGMOID><<<dim3(M / 64, 2), blk, 0, stream>>>(iproj, Wk, bk, Kb, nullptr, nullptr, nullptr, M, 256, 256);
    gemm_bf16<MODE_NONE><<<dim3(M / 64, 2), blk, 0, stream>>>(iproj, Wv, bv, Vb, nullptr, nullptr, nullptr, M, 256, 256);
    fuzzy_attn<<<dim3(32, 16), 128, 0, stream>>>(Qb, Kb, Vb, Ob);
    gemm_bf16<MODE_NONE><<<dim3(M / 64, 2), blk, 0, stream>>>(Ob, Wo, bo, E1, nullptr, nullptr, nullptr, M, 256, 256);

    // attention 2: image queries, text keys/values (buffers reused)
    gemm_bf16<MODE_SIGMOID><<<dim3(M / 64, 2), blk, 0, stream>>>(iproj, Wq, bq, Qb, nullptr, nullptr, nullptr, M, 256, 256);
    gemm_bf16<MODE_SIGMOID><<<dim3(M / 64, 2), blk, 0, stream>>>(tproj, Wk, bk, Kb, nullptr, nullptr, nullptr, M, 256, 256);
    gemm_bf16<MODE_NONE><<<dim3(M / 64, 2), blk, 0, stream>>>(tproj, Wv, bv, Vb, nullptr, nullptr, nullptr, M, 256, 256);
    fuzzy_attn<<<dim3(32, 16), 128, 0, stream>>>(Qb, Kb, Vb, Ob);
    gemm_bf16<MODE_NONE><<<dim3(M / 64, 2), blk, 0, stream>>>(Ob, Wo, bo, E2, nullptr, nullptr, nullptr, M, 256, 256);

    // output projections + fusion blend (f32 epilogue)
    gemm_bf16<MODE_RESID><<<dim3(M / 64, 4), blk, 0, stream>>>(E1, Wto, bto, nullptr, outT, tf,  fw, M, 512, 256);
    gemm_bf16<MODE_RESID><<<dim3(M / 64, 6), blk, 0, stream>>>(E2, Wio, bio, nullptr, outI, imf, fw, M, 768, 256);
}
